// EdgeConvLayer_39737037423416
// MI455X (gfx1250) — matrix operand order repaired and checked by reading, not compiled, not run on hardware
//
#include <hip/hip_runtime.h>
#include <hip/hip_bf16.h>

typedef __attribute__((ext_vector_type(16))) _Float16 v16h;
typedef __attribute__((ext_vector_type(8)))  float    v8f;

#define DIM 96
#define ED  17
#define KP  32   // K padded for layer-1 (17 -> 32)

// ---------------------------------------------------------------------------
// Fragment loaders per CDNA5 ISA 7.12.2 (wave32, 16-bit operands)
// A 16x32 (MxK): lane 0-15 row m, VGPR v<4 holds K=2v..2v+1 (+8 for lanes
// 16-31); v>=4 holds K=16+2(v-4)... (+8 for upper half).
// ---------------------------------------------------------------------------
__device__ __forceinline__ v16h load_a_frag(const _Float16* a, int stride,
                                            int mrow, int half) {
  v16h r;
#pragma unroll
  for (int j = 0; j < 16; ++j) {
    int v = j >> 1, t = j & 1;
    int k = ((v < 4) ? (2 * v) : (16 + 2 * (v - 4))) + 8 * half + t;
    r[j] = a[mrow * stride + k];
  }
  return r;
}

// B 32x16 (KxN): lane 0-15 column n, same K layout as A (v<4: K=2v..2v+1,
// v>=4: K=16+2(v-4).., +8 for lanes 16-31).  Weights staged in LDS as W[n][k] so B[k][n] = W[n][k].
__device__ __forceinline__ v16h load_b_frag(const _Float16* w, int stride,
                                            int mrow, int half) {
  v16h r;
#pragma unroll
  for (int j = 0; j < 16; ++j) {
    int v = j >> 1, t = j & 1;
    int k = ((v < 4) ? (2 * v) : (16 + 2 * (v - 4))) + 8 * half + t;
    r[j] = w[mrow * stride + k];
  }
  return r;
}

// ---------------------------------------------------------------------------
// Kernel 0: zero the agg accumulator (d_ws) — replayed every launch.
// ---------------------------------------------------------------------------
__global__ void zero_kernel(float* __restrict__ p, long long n) {
  long long i = (long long)blockIdx.x * blockDim.x + threadIdx.x;
  long long stride = (long long)gridDim.x * blockDim.x;
  for (; i < n; i += stride) p[i] = 0.0f;
}

// ---------------------------------------------------------------------------
// Kernel 1: edge MLP (2 WMMA GEMM stages) + gather x[src] + scatter-add agg.
// 256 threads = 8 waves; each wave owns a 16-edge x 96-col tile.
// ---------------------------------------------------------------------------
__global__ __launch_bounds__(256) void edge_kernel(
    const float* __restrict__ x, const long long* __restrict__ ei,
    const float* __restrict__ ea, const float* __restrict__ W1,
    const float* __restrict__ b1, const float* __restrict__ W2,
    const float* __restrict__ b2, float* __restrict__ agg,
    int N, int E) {
  __shared__ _Float16 sW1[DIM][KP];     //  6.0 KB  W1 K-padded, [n][k]
  __shared__ _Float16 sW2[DIM][DIM];    // 18.0 KB  W2 [n][k]
  __shared__ float    sb1[DIM];
  __shared__ float    sb2[DIM];
  __shared__ _Float16 sEA[128][KP];     //  8.0 KB  edge_attr tile, K-padded
  __shared__ _Float16 sH[128][DIM];     // 24.0 KB  hidden activations (f16)
  __shared__ int      sSrc[128];
  __shared__ int      sDst[128];

  const int tid  = threadIdx.x;
  const int lane = tid & 31;
  const int wv   = tid >> 5;       // wave id 0..7
  const int half = lane >> 4;
  const int mrow = lane & 15;

  // ---- stage weights / biases (block-cooperative) ----
  for (int i = tid; i < DIM * KP; i += 256) {
    int n = i / KP, k = i % KP;
    sW1[n][k] = (k < ED) ? (_Float16)W1[n * ED + k] : (_Float16)0.0f;
  }
  for (int i = tid; i < DIM * DIM; i += 256)
    sW2[i / DIM][i % DIM] = (_Float16)W2[i];
  for (int i = tid; i < DIM; i += 256) { sb1[i] = b1[i]; sb2[i] = b2[i]; }

  // ---- stage 128-edge tile of edge_attr + indices ----
  const int eblk = blockIdx.x * 128;
  for (int i = tid; i < 128 * KP; i += 256) {
    int m = i >> 5, k = i & 31;
    int e = eblk + m;
    sEA[m][k] = (k < ED && e < E) ? (_Float16)ea[e * ED + k] : (_Float16)0.0f;
  }
  for (int i = tid; i < 128; i += 256) {
    int e = eblk + i;
    sSrc[i] = (e < E) ? (int)ei[e] : 0;
    sDst[i] = (e < E) ? (int)ei[(long long)E + e] : 0;
  }
  __syncthreads();

  // ---- layer 1: h = relu(edge_attr @ W1.T + b1), 16x96 per wave ----
  v16h a1 = load_a_frag(&sEA[wv * 16][0], KP, mrow, half);
#pragma unroll
  for (int nt = 0; nt < 6; ++nt) {
    v16h bf = load_b_frag(&sW1[nt * 16][0], KP, mrow, half);
    v8f acc = {};
    acc = __builtin_amdgcn_wmma_f32_16x16x32_f16(false, a1, false, bf,
                                                 (short)0, acc, false, false);
    float bias = sb1[nt * 16 + mrow];
#pragma unroll
    for (int r = 0; r < 8; ++r) {
      float h = acc[r] + bias;
      h = h > 0.0f ? h : 0.0f;
      sH[wv * 16 + r + 8 * half][nt * 16 + mrow] = (_Float16)h;
    }
  }
  __syncthreads();

  // ---- layer 2: w = h @ W2.T + b2 (K = 96), then gather/mul/scatter ----
  v16h a2[3];
#pragma unroll
  for (int kc = 0; kc < 3; ++kc)
    a2[kc] = load_a_frag(&sH[wv * 16][kc * 32], DIM, mrow, half);

#pragma unroll
  for (int nt = 0; nt < 6; ++nt) {
    v8f acc = {};
#pragma unroll
    for (int kc = 0; kc < 3; ++kc) {
      v16h bf = load_b_frag(&sW2[nt * 16][kc * 32], DIM, mrow, half);
      acc = __builtin_amdgcn_wmma_f32_16x16x32_f16(false, a2[kc], false, bf,
                                                   (short)0, acc, false, false);
    }
    const int n = nt * 16 + mrow;
    const float bias = sb2[n];
#pragma unroll
    for (int r = 0; r < 8; ++r) {
      int le = wv * 16 + r + 8 * half;     // local edge in block
      int e  = eblk + le;
      if (e < E) {
        float w   = acc[r] + bias;
        float msg = w * __ldg(&x[(long long)sSrc[le] * DIM + n]); // L2-resident gather
        atomicAdd(&agg[(long long)sDst[le] * DIM + n], msg);      // L2-resident scatter
      }
    }
  }
}

// ---------------------------------------------------------------------------
// Kernel 2: out = relu([x|agg] @ [Ws|Wn].T + (bs+bn)) as one K=192 WMMA GEMM.
// 128 threads = 4 waves; each wave owns a 16-node x 96-col tile.
// ---------------------------------------------------------------------------
__global__ __launch_bounds__(128) void node_kernel(
    const float* __restrict__ x, const float* __restrict__ agg,
    const float* __restrict__ Ws, const float* __restrict__ bs,
    const float* __restrict__ Wn, const float* __restrict__ bn,
    float* __restrict__ out, int N) {
  __shared__ _Float16 sW[DIM][192];      // 36 KB  [Ws|Wn] as [n][k]
  __shared__ float    sB[DIM];
  __shared__ _Float16 sXA[4][16][192];   // 24 KB  [x|agg] tiles (f16)

  const int tid  = threadIdx.x;
  const int lane = tid & 31;
  const int wv   = tid >> 5;       // 0..3
  const int half = lane >> 4;
  const int mrow = lane & 15;

  for (int i = tid; i < DIM * 192; i += 128) {
    int n = i / 192, k = i % 192;
    float v = (k < DIM) ? Ws[n * DIM + k] : Wn[n * DIM + (k - DIM)];
    sW[n][k] = (_Float16)v;
  }
  for (int i = tid; i < DIM; i += 128) sB[i] = bs[i] + bn[i];

  for (int i = tid; i < 4 * 16 * 192; i += 128) {
    int w = i / (16 * 192), rem = i % (16 * 192);
    int m = rem / 192, k = rem % 192;
    int node = (blockIdx.x * 4 + w) * 16 + m;
    float v = 0.0f;
    if (node < N)
      v = (k < DIM) ? x[(long long)node * DIM + k]
                    : agg[(long long)node * DIM + (k - DIM)];
    sXA[w][m][k] = (_Float16)v;
  }
  __syncthreads();

  v16h af[6];
#pragma unroll
  for (int kc = 0; kc < 6; ++kc)
    af[kc] = load_a_frag(&sXA[wv][0][kc * 32], 192, mrow, half);

  const int nb = (blockIdx.x * 4 + wv) * 16;
#pragma unroll
  for (int nt = 0; nt < 6; ++nt) {
    v8f acc = {};
#pragma unroll
    for (int kc = 0; kc < 6; ++kc) {
      v16h bf = load_b_frag(&sW[nt * 16][kc * 32], 192, mrow, half);
      acc = __builtin_amdgcn_wmma_f32_16x16x32_f16(false, af[kc], false, bf,
                                                   (short)0, acc, false, false);
    }
    const int n = nt * 16 + mrow;
    const float bias = sB[n];
#pragma unroll
    for (int r = 0; r < 8; ++r) {
      int node = nb + r + 8 * half;
      if (node < N) {
        float v = acc[r] + bias;
        out[(long long)node * DIM + n] = v > 0.0f ? v : 0.0f;
      }
    }
  }
}

// ---------------------------------------------------------------------------
// inputs: x, edge_index, edge_attr, W1, b1, W2, b2, Ws, bs, Wn, bn
// ---------------------------------------------------------------------------
extern "C" void kernel_launch(void* const* d_in, const int* in_sizes, int n_in,
                              void* d_out, int out_size, void* d_ws,
                              size_t ws_size, hipStream_t stream) {
  const float*     x   = (const float*)d_in[0];
  const long long* ei  = (const long long*)d_in[1];   // int64 [2, E]
  const float*     ea  = (const float*)d_in[2];
  const float*     W1  = (const float*)d_in[3];
  const float*     b1  = (const float*)d_in[4];
  const float*     W2  = (const float*)d_in[5];
  const float*     b2  = (const float*)d_in[6];
  const float*     Ws  = (const float*)d_in[7];
  const float*     bs  = (const float*)d_in[8];
  const float*     Wn  = (const float*)d_in[9];
  const float*     bn  = (const float*)d_in[10];
  float*           out = (float*)d_out;

  const int N = in_sizes[0] / DIM;
  const int E = in_sizes[2] / ED;

  float* agg = (float*)d_ws;   // N*DIM floats of scratch

  // 1) zero accumulator (must be re-done every replay)
  zero_kernel<<<1024, 256, 0, stream>>>(agg, (long long)N * DIM);

  // 2) edge MLP + gather + scatter-add (128 edges per block)
  edge_kernel<<<(E + 127) / 128, 256, 0, stream>>>(x, ei, ea, W1, b1, W2, b2,
                                                   agg, N, E);

  // 3) fused node GEMMs + bias + ReLU (64 nodes per block)
  int ntiles = (N + 15) / 16;
  node_kernel<<<(ntiles + 3) / 4, 128, 0, stream>>>(x, agg, Ws, bs, Wn, bn,
                                                    out, N);
}
